// EfficientWindowAttention_17609365914261
// MI455X (gfx1250) — matrix operand order repaired and checked by reading, not compiled, not run on hardware
//
#include <hip/hip_runtime.h>

// ---------------- vector types ----------------
typedef __attribute__((ext_vector_type(16))) __bf16        v16bf;
typedef __attribute__((ext_vector_type(8)))  float         v8f;
typedef __attribute__((ext_vector_type(4)))  unsigned int  u32x4;
typedef __attribute__((ext_vector_type(2)))  unsigned int  u32x2;
typedef __attribute__((ext_vector_type(8)))  unsigned int  u32x8;
typedef __attribute__((ext_vector_type(4)))  float         f32x4;

// ---------------- helpers ----------------
__device__ __forceinline__ unsigned short f2bf_u(float f) {
  unsigned int u = __builtin_bit_cast(unsigned int, f);
  u += 0x7FFFu + ((u >> 16) & 1u);            // round-to-nearest-even
  return (unsigned short)(u >> 16);
}
__device__ __forceinline__ unsigned int pk2bf(float a, float b) {
  return (unsigned int)f2bf_u(a) | ((unsigned int)f2bf_u(b) << 16);
}
__device__ __forceinline__ v8f wmma_bf16(v16bf a, v16bf b, v8f c) {
  return __builtin_amdgcn_wmma_f32_16x16x32_bf16(false, a, false, b, (short)0, c, false, false);
}

// A-fragment: 16x32 bf16, memory row-major [m][k], stride in halfs.
// lane L: row = mBase + L%16 ; halfs [kb..kb+7] and [kb+16..kb+23], kb = kBase + 8*(L/16)
__device__ __forceinline__ v16bf loadA(const __bf16* base, int strideH, int mBase, int kBase, int lane) {
  int r = mBase + (lane & 15);
  int k = kBase + ((lane >> 4) << 3);
  const __bf16* p = base + r * strideH + k;
  u32x4 lo = *(const u32x4*)p;
  u32x4 hi = *(const u32x4*)(p + 16);
  u32x8 t = __builtin_shufflevector(lo, hi, 0, 1, 2, 3, 4, 5, 6, 7);
  return __builtin_bit_cast(v16bf, t);
}
// B-fragment: 32x16 bf16 (KxN), memory stored K-contiguous per column: [n][k], stride in halfs.
// lane L: col = cBase + L%16 ; halfs [kb..kb+7] and [kb+16..kb+23], kb = kBase + 8*(L/16)
__device__ __forceinline__ v16bf loadB(const __bf16* base, int strideH, int cBase, int kBase, int lane) {
  int c = cBase + (lane & 15);
  int k = kBase + ((lane >> 4) << 3);
  const __bf16* p = base + c * strideH + k;
  u32x4 lo = *(const u32x4*)p;
  u32x4 hi = *(const u32x4*)(p + 16);
  u32x8 t = __builtin_shufflevector(lo, hi, 0, 1, 2, 3, 4, 5, 6, 7);
  return __builtin_bit_cast(v16bf, t);
}
// D-tile transposed store: writes T[c][m] contiguously (16B per lane)
__device__ __forceinline__ void storeDT(__bf16* base, int strideH, int cBase, int mBase, v8f a, int lane) {
  int c = cBase + (lane & 15);
  int m = mBase + ((lane >> 4) << 3);
  u32x4 v = { pk2bf(a[0], a[1]), pk2bf(a[2], a[3]), pk2bf(a[4], a[5]), pk2bf(a[6], a[7]) };
  *(u32x4*)(base + c * strideH + m) = v;
}
// D-tile row-major scatter store (8 x b16 per lane)
__device__ __forceinline__ void storeDRM(__bf16* base, int strideH, int cBase, int mBase, v8f a, int lane) {
  int c = cBase + (lane & 15);
  int mb = mBase + ((lane >> 4) << 3);
#pragma unroll
  for (int j = 0; j < 8; ++j)
    base[(mb + j) * strideH + c] = __builtin_bit_cast(__bf16, f2bf_u(a[j]));
}

// QKV GEMM block: [64 rows] x [32 cols at colG], K=512 from lx (LDS bf16, stride 520).
// ks-outer with 8 accumulators: each A fragment feeds 2 WMMAs; B fragments are
// software-pipelined one k-step ahead to hide global latency under the WMMAs.
__device__ __forceinline__ void qkv_block(const __bf16* lx, const __bf16* wqkvT, const float* b_qkv,
                                          int colG, int lane, bool transposed, __bf16* dst, int dstStride) {
  v8f acc[2][4];
#pragma unroll
  for (int ct = 0; ct < 2; ++ct)
#pragma unroll
    for (int rt = 0; rt < 4; ++rt) acc[ct][rt] = (v8f){0, 0, 0, 0, 0, 0, 0, 0};

  v16bf b0 = loadB(wqkvT, 512, colG, 0, lane);
  v16bf b1 = loadB(wqkvT, 512, colG + 16, 0, lane);
#pragma unroll
  for (int ks = 0; ks < 16; ++ks) {
    int ksn = ((ks + 1) & 15) * 32;               // wrap: last prefetch is harmless
    v16bf nb0 = loadB(wqkvT, 512, colG, ksn, lane);
    v16bf nb1 = loadB(wqkvT, 512, colG + 16, ksn, lane);
#pragma unroll
    for (int rt = 0; rt < 4; ++rt) {
      v16bf a = loadA(lx, 520, rt * 16, ks * 32, lane);
      acc[0][rt] = wmma_bf16(a, b0, acc[0][rt]);
      acc[1][rt] = wmma_bf16(a, b1, acc[1][rt]);
    }
    b0 = nb0;
    b1 = nb1;
  }
#pragma unroll
  for (int ct = 0; ct < 2; ++ct) {
    float bb = b_qkv[colG + ct * 16 + (lane & 15)];
#pragma unroll
    for (int rt = 0; rt < 4; ++rt) {
#pragma unroll
      for (int j = 0; j < 8; ++j) acc[ct][rt][j] += bb;
      if (transposed) storeDT(dst, dstStride, ct * 16, rt * 16, acc[ct][rt], lane);
      else            storeDRM(dst, dstStride, ct * 16, rt * 16, acc[ct][rt], lane);
    }
  }
}

// ---------------- prep kernel: repack weights (bf16, transposed) + bias gather ----------------
__global__ __launch_bounds__(256) void ewa_prep(const float* __restrict__ w_qkv, const float* __restrict__ w_proj,
                                                const float* __restrict__ w_k, const float* __restrict__ w_v,
                                                const float* __restrict__ bias_table,
                                                __bf16* __restrict__ wqkvT, __bf16* __restrict__ wprojT,
                                                __bf16* __restrict__ wkB, __bf16* __restrict__ wvB,
                                                float* __restrict__ biasM) {
  int idx = blockIdx.x * 256 + threadIdx.x;
  if (idx < 786432) {  // wqkvT[c][k] = w_qkv[k][c]
    int c = idx >> 9, k = idx & 511;
    wqkvT[idx] = __builtin_bit_cast(__bf16, f2bf_u(w_qkv[k * 1536 + c]));
    return;
  }
  idx -= 786432;
  if (idx < 262144) {  // wprojT[c][k] = w_proj[k][c]
    int c = idx >> 9, k = idx & 511;
    wprojT[idx] = __builtin_bit_cast(__bf16, f2bf_u(w_proj[k * 512 + c]));
    return;
  }
  idx -= 262144;
  if (idx < 1024) { wkB[idx] = __builtin_bit_cast(__bf16, f2bf_u(w_k[idx])); return; }
  idx -= 1024;
  if (idx < 1024) { wvB[idx] = __builtin_bit_cast(__bf16, f2bf_u(w_v[idx])); return; }
  idx -= 1024;
  if (idx < 16384) {   // biasM[h][n][p] = bias_table[rel(n,p)][h]
    int hq = idx >> 10;
    int r  = (idx >> 4) & 63;
    int pp = idx & 15;
    int iy = r >> 3, ix = r & 7, jy = pp >> 3, jx = pp & 7;
    int rel = (iy - jy + 7) * 15 + (ix - jx + 7);
    biasM[idx] = bias_table[rel * 16 + hq];
  }
}

// ---------------- fused per-window attention kernel ----------------
// smem layout (bytes):
//   lx  [64][520] bf16      @ 0       (66560)
//   ao  [64][520] bf16      @ 66560   (66560)
//   per wave w @ 133120 + w*19200:
//     kT  [32][72] bf16  +0     (4608)   k^T  (d rows, n contiguous)
//     vT  [32][72] bf16  +4608  (4608)   v^T
//     kp  [16][40] bf16  +9216  (1280)   scale * (w_k @ k)  [p][d]
//     vpT [32][40] bf16  +10496 (2560)   (w_v @ v)^T  [d][p], p>=16 zero
//     qb  [64][40] bf16  +13056 (5120)   q + bias     [n][d]
//     at  [16][32] bf16  +18176 (1024)   softmax(attn) rows, p>=16 zero
__global__ __launch_bounds__(256) void ewa_fused(const float* __restrict__ x,
                                                 const float* __restrict__ b_qkv,
                                                 const float* __restrict__ b_proj,
                                                 const __bf16* __restrict__ wqkvT,
                                                 const __bf16* __restrict__ wprojT,
                                                 const __bf16* __restrict__ wkB,
                                                 const __bf16* __restrict__ wvB,
                                                 const float* __restrict__ biasM,
                                                 float* __restrict__ out) {
  extern __shared__ __attribute__((aligned(16))) char smem[];
  __bf16* lx = (__bf16*)smem;
  __bf16* ao = (__bf16*)(smem + 66560);
  int tid = threadIdx.x, lane = tid & 31, wid = tid >> 5;
  char* wb = smem + 133120 + wid * 19200;
  __bf16* kT  = (__bf16*)(wb);
  __bf16* vT  = (__bf16*)(wb + 4608);
  __bf16* kp  = (__bf16*)(wb + 9216);
  __bf16* vpT = (__bf16*)(wb + 10496);
  __bf16* qb  = (__bf16*)(wb + 13056);
  __bf16* at  = (__bf16*)(wb + 18176);

  // ---- phase 0: x window f32 -> bf16 LDS ----
  const f32x4* x4 = (const f32x4*)(x + (size_t)blockIdx.x * 32768);
#pragma unroll
  for (int i = 0; i < 32; ++i) {
    int e4 = tid + i * 256;
    f32x4 d = x4[e4];
    int row = e4 >> 7;
    int col = (e4 & 127) << 2;
    u32x2 pv = { pk2bf(d[0], d[1]), pk2bf(d[2], d[3]) };
    *(u32x2*)(lx + row * 520 + col) = pv;
  }
  __syncthreads();

  const float scale = 0.17677669529663687f;  // 32^-0.5 (folded into kp)
  int p = lane & 15, hi8 = (lane >> 4) << 3;

  // ---- per-head fused QKV + Linformer attention (2 heads per wave) ----
  for (int hh = 0; hh < 2; ++hh) {
    int h = wid * 2 + hh;
    qkv_block(lx, wqkvT, b_qkv, 512 + h * 32, lane, true, kT, 72);   // K -> kT[d][n]
    qkv_block(lx, wqkvT, b_qkv, 1024 + h * 32, lane, true, vT, 72);  // V -> vT[d][n]
    qkv_block(lx, wqkvT, b_qkv, h * 32, lane, false, qb, 40);        // Q -> qb[n][d]

    // kp[p][d] = scale * sum_n w_k[p][n] k[n][d]  (computed as kp^T = kT x w_k, stored transposed)
#pragma unroll
    for (int mt = 0; mt < 2; ++mt) {
      v8f acc = (v8f){0, 0, 0, 0, 0, 0, 0, 0};
#pragma unroll
      for (int kk = 0; kk < 2; ++kk) {
        v16bf a = loadA(kT, 72, mt * 16, kk * 32, lane);
        v16bf b = loadB(wkB, 64, 0, kk * 32, lane);
        acc = wmma_bf16(a, b, acc);
      }
#pragma unroll
      for (int j = 0; j < 8; ++j) acc[j] *= scale;
      storeDT(kp, 40, 0, mt * 16, acc, lane);
    }

    // zero vpT (need p>=16 zero for K=32 padding)
    {
      u32x4* z = (u32x4*)(vpT + lane * 40);
#pragma unroll
      for (int j = 0; j < 5; ++j) z[j] = (u32x4){0, 0, 0, 0};
    }
    // vp[p][d] = sum_n w_v[p][n] v[n][d] ; stored transposed -> vpT[d][p]
#pragma unroll
    for (int ct = 0; ct < 2; ++ct) {
      v8f acc = (v8f){0, 0, 0, 0, 0, 0, 0, 0};
#pragma unroll
      for (int kk = 0; kk < 2; ++kk) {
        v16bf a = loadA(wvB, 64, 0, kk * 32, lane);
        v16bf b = loadB(vT, 72, ct * 16, kk * 32, lane);
        acc = wmma_bf16(a, b, acc);
      }
      storeDT(vpT, 40, ct * 16, 0, acc, lane);
    }

    // attention: per 16-row tile of q
    for (int rt = 0; rt < 4; ++rt) {
      {  // zero attn buffer (p>=16 padding)
        u32x4* z = (u32x4*)(at + lane * 16);
        z[0] = (u32x4){0, 0, 0, 0};
        z[1] = (u32x4){0, 0, 0, 0};
      }
      v8f acc = (v8f){0, 0, 0, 0, 0, 0, 0, 0};
      {
        v16bf a = loadA(qb, 40, rt * 16, 0, lane);
        v16bf b = loadB(kp, 40, 0, 0, lane);
        acc = wmma_bf16(a, b, acc);
      }
      // softmax over p (16 lanes share a row) with rel-pos bias
      float vals[8];
      float mx = -3.402823466e38f;
#pragma unroll
      for (int j = 0; j < 8; ++j) {
        int n = rt * 16 + hi8 + j;
        vals[j] = acc[j] + biasM[(h * 64 + n) * 16 + p];
        mx = fmaxf(mx, vals[j]);
      }
      mx = fmaxf(mx, __shfl_xor(mx, 1, 32));
      mx = fmaxf(mx, __shfl_xor(mx, 2, 32));
      mx = fmaxf(mx, __shfl_xor(mx, 4, 32));
      mx = fmaxf(mx, __shfl_xor(mx, 8, 32));
      float s = 0.f;
#pragma unroll
      for (int j = 0; j < 8; ++j) { vals[j] = __expf(vals[j] - mx); s += vals[j]; }
      s += __shfl_xor(s, 1, 32);
      s += __shfl_xor(s, 2, 32);
      s += __shfl_xor(s, 4, 32);
      s += __shfl_xor(s, 8, 32);
      float inv = 1.0f / s;
#pragma unroll
      for (int j = 0; j < 8; ++j)
        at[(hi8 + j) * 32 + p] = __builtin_bit_cast(__bf16, f2bf_u(vals[j] * inv));

      // out^T tile: vpT[d][p] x at[n][p]  -> D[m=d][c=n]; transposed-store -> ao[n][h*32+d]
#pragma unroll
      for (int mt = 0; mt < 2; ++mt) {
        v16bf a = loadA(vpT, 40, mt * 16, 0, lane);
        v16bf b = loadB(at, 32, 0, 0, lane);
        v8f o = (v8f){0, 0, 0, 0, 0, 0, 0, 0};
        o = wmma_bf16(a, b, o);
        storeDT(ao, 520, rt * 16, h * 32 + mt * 16, o, lane);
      }
    }
  }
  __syncthreads();

  // ---- output projection: Y = ao[64,512] @ w_proj + b_proj ----
  // Pair the wave's col tiles so each A fragment feeds 2 WMMAs; pipeline B loads.
  float* ob = out + (size_t)blockIdx.x * 32768;
#pragma unroll
  for (int cp = 0; cp < 2; ++cp) {
    int c0 = (wid + cp * 16) * 16;
    int c1 = (wid + 8 + cp * 16) * 16;
    v8f acc[2][4];
#pragma unroll
    for (int ct = 0; ct < 2; ++ct)
#pragma unroll
      for (int rt = 0; rt < 4; ++rt) acc[ct][rt] = (v8f){0, 0, 0, 0, 0, 0, 0, 0};

    v16bf b0 = loadB(wprojT, 512, c0, 0, lane);
    v16bf b1 = loadB(wprojT, 512, c1, 0, lane);
#pragma unroll
    for (int ks = 0; ks < 16; ++ks) {
      int ksn = ((ks + 1) & 15) * 32;
      v16bf nb0 = loadB(wprojT, 512, c0, ksn, lane);
      v16bf nb1 = loadB(wprojT, 512, c1, ksn, lane);
#pragma unroll
      for (int rt = 0; rt < 4; ++rt) {
        v16bf a = loadA(ao, 520, rt * 16, ks * 32, lane);
        acc[0][rt] = wmma_bf16(a, b0, acc[0][rt]);
        acc[1][rt] = wmma_bf16(a, b1, acc[1][rt]);
      }
      b0 = nb0;
      b1 = nb1;
    }
#pragma unroll
    for (int ct = 0; ct < 2; ++ct) {
      int cB = ct ? c1 : c0;
      float bb = b_proj[cB + p];
#pragma unroll
      for (int rt = 0; rt < 4; ++rt)
#pragma unroll
        for (int j = 0; j < 8; ++j)
          ob[(size_t)(rt * 16 + hi8 + j) * 512 + cB + p] = acc[ct][rt][j] + bb;
    }
  }
}

// ---------------- launch ----------------
extern "C" void kernel_launch(void* const* d_in, const int* in_sizes, int n_in,
                              void* d_out, int out_size, void* d_ws, size_t ws_size,
                              hipStream_t stream) {
  const float* x          = (const float*)d_in[0];
  const float* w_qkv      = (const float*)d_in[1];
  const float* b_qkv      = (const float*)d_in[2];
  const float* w_proj     = (const float*)d_in[3];
  const float* b_proj     = (const float*)d_in[4];
  const float* w_k        = (const float*)d_in[5];
  const float* w_v        = (const float*)d_in[6];
  const float* bias_table = (const float*)d_in[7];

  char* ws = (char*)d_ws;
  __bf16* wqkvT  = (__bf16*)(ws);            // 1,572,864 B
  __bf16* wprojT = (__bf16*)(ws + 1572864);  //   524,288 B
  __bf16* wkB    = (__bf16*)(ws + 2097152);  //     2,048 B
  __bf16* wvB    = (__bf16*)(ws + 2099200);  //     2,048 B
  float*  biasM  = (float*)(ws + 2101248);   //    65,536 B

  ewa_prep<<<4168, 256, 0, stream>>>(w_qkv, w_proj, w_k, w_v, bias_table,
                                     wqkvT, wprojT, wkB, wvB, biasM);
  ewa_fused<<<2048, 256, 286720, stream>>>(x, b_qkv, b_proj, wqkvT, wprojT,
                                           wkB, wvB, biasM, (float*)d_out);
}
